// PhantomGateLayer_47717086659264
// MI455X (gfx1250) — matrix operand order repaired and checked by reading, not compiled, not run on hardware
//
#include <hip/hip_runtime.h>
#include <cstdint>

#define BB   2
#define LL   1024
#define HIDD 1024
#define NHD  16
#define DH   64
#define TOK  (BB*LL)   // 2048

typedef __attribute__((ext_vector_type(16))) __bf16 v16bf;
typedef __attribute__((ext_vector_type(8)))  __bf16 v8bf;
typedef __attribute__((ext_vector_type(8)))  float  v8f;
typedef __attribute__((ext_vector_type(4)))  unsigned int u32x4;
typedef __attribute__((ext_vector_type(8)))  int          i32x8;
typedef __attribute__((ext_vector_type(4)))  int          i32x4;

__device__ __forceinline__ unsigned short f2bfbits(float f) {
  unsigned u = __builtin_bit_cast(unsigned, f);
  u = (u + 0x7FFFu + ((u >> 16) & 1u)) >> 16;   // round-to-nearest-even
  return (unsigned short)u;
}
__device__ __forceinline__ float sigm(float x) {
  return __builtin_amdgcn_rcpf(1.0f + __expf(-x));   // v_rcp_f32, no div sequence
}

// ---------------------------------------------------------------------------
// One-shot f32 -> bf16 conversion (vectorized: float4 in, 4x bf16 out)
// ---------------------------------------------------------------------------
__global__ __launch_bounds__(256)
void to_bf16(const float* __restrict__ in, unsigned short* __restrict__ out, int n4)
{
  const int i = blockIdx.x * 256 + threadIdx.x;
  if (i >= n4) return;
  const float4 f = ((const float4*)in)[i];
  union { unsigned short s[4]; uint2 u; } r;
  r.s[0] = f2bfbits(f.x); r.s[1] = f2bfbits(f.y);
  r.s[2] = f2bfbits(f.z); r.s[3] = f2bfbits(f.w);
  ((uint2*)out)[i] = r.u;
}

// ---------------------------------------------------------------------------
// TDM: DMA one 64-row x 64-col bf16 tile (row stride Kdim elements) from
// global to LDS, inserting 16B of LDS padding after every 128B row so LDS
// rows land on a 144B (72-element) pitch.  D# built per CDNA5 ISA.
// This toolchain exposes the 6-arg builtin: (g0 u32x4, g1 i32x8, g2 i32x4,
// g3 i32x4, i32x8, cpol) — trailing groups zero for a plain 2D tile.
// ---------------------------------------------------------------------------
__device__ __forceinline__ void tdm_load_tile_2d(unsigned lds_byte_off,
                                                 const void* gaddr, int Kdim)
{
  const unsigned long long ga = (unsigned long long)(uintptr_t)gaddr;
  u32x4 g0;
  g0[0] = 1u;                                        // count=1 (valid), user mode
  g0[1] = lds_byte_off;                              // lds_addr  (bits 63:32)
  g0[2] = (unsigned)ga;                              // global_addr[31:0]  (95:64)
  g0[3] = (unsigned)((ga >> 32) & 0x01FFFFFFu)       // global_addr[56:32] (120:96)
        | (2u << 30);                                // type=2 ("image")
  i32x8 g1;
  g1[0] = (1 << 16)                                  // data_size = 2 bytes
        | (1 << 20)                                  // pad_enable
        | (4 << 22)                                  // pad_interval: 16x8B = 128B
        | (3 << 25);                                 // pad_amount: 4 DWORDs = 16B
  g1[1] = (Kdim & 0xFFFF) << 16;                     // tensor_dim0 lo16 (63:48)
  g1[2] = (int)(((unsigned)(Kdim >> 16) & 0xFFFFu)   // tensor_dim0 hi16 (79:64)
        | (0xFFFFu << 16));                          // tensor_dim1 lo16 = 65535
  g1[3] = (64 << 16);                                // tensor_dim1 hi=0, tile_dim0=64
  g1[4] = 64;                                        // tile_dim1=64, tile_dim2=0
  g1[5] = Kdim;                                      // tensor_dim0_stride lo32
  g1[6] = 0;                                         // stride hi / dim1_stride lo
  g1[7] = 0;
  const i32x4 z4 = {0, 0, 0, 0};                     // groups 2/3 unused (2D)
  const i32x8 z8 = {0, 0, 0, 0, 0, 0, 0, 0};
  __builtin_amdgcn_tensor_load_to_lds(g0, g1, z4, z4, z8, 0);
}

// ---------------------------------------------------------------------------
// Y[M x N] = act( X[MxK] @ W[NxK]^T + bias[N] ), X/W pre-converted bf16.
// Block = 128 thr (4 wave32, 2x2), tile 64x64, K-step 64 (8 WMMA / stage).
// TDM double-buffered staging: wave0 issues 2 tensor_load_to_lds one stage
// ahead, s_wait_tensorcnt 2 retires the current stage before the barrier.
// ---------------------------------------------------------------------------
__global__ __launch_bounds__(128)
void gemm_bf16_wmma(const unsigned short* __restrict__ X,
                    const unsigned short* __restrict__ W,
                    const float* __restrict__ bias, float* __restrict__ Y,
                    int M, int N, int Kdim, int act)
{
  __shared__ __align__(16) __bf16 sA[2][64][72];   // 144B pitch (TDM-padded)
  __shared__ __align__(16) __bf16 sB[2][64][72];

  const int tid  = threadIdx.x;
  const int lane = tid & 31;
  const int wave = tid >> 5;
  const int g    = lane >> 4;
  const int hl   = lane & 15;
  const int wm   = (wave >> 1) * 32;
  const int wn   = (wave & 1)  * 32;
  const int Mbase = blockIdx.y * 64;
  const int Nbase = blockIdx.x * 64;

  const unsigned short* Xt = X + (size_t)Mbase * Kdim;
  const unsigned short* Wt = W + (size_t)Nbase * Kdim;

  v8f acc[2][2];
  #pragma unroll
  for (int i = 0; i < 2; ++i)
    #pragma unroll
    for (int j = 0; j < 2; ++j)
      acc[i][j] = (v8f){0.f,0.f,0.f,0.f,0.f,0.f,0.f,0.f};

  const unsigned ldsA[2] = { (unsigned)(uintptr_t)&sA[0][0][0],
                             (unsigned)(uintptr_t)&sA[1][0][0] };
  const unsigned ldsB[2] = { (unsigned)(uintptr_t)&sB[0][0][0],
                             (unsigned)(uintptr_t)&sB[1][0][0] };

  const int nk = Kdim >> 6;
  if (wave == 0) {                     // waves 1-3: EXEC=0 & TENSORcnt=0 -> skipped
    tdm_load_tile_2d(ldsA[0], Xt, Kdim);
    tdm_load_tile_2d(ldsB[0], Wt, Kdim);
  }

  for (int ki = 0; ki < nk; ++ki) {
    const int buf = ki & 1;
    if (wave == 0) {
      if (ki + 1 < nk) {
        tdm_load_tile_2d(ldsA[buf ^ 1], Xt + (size_t)(ki + 1) * 64, Kdim);
        tdm_load_tile_2d(ldsB[buf ^ 1], Wt + (size_t)(ki + 1) * 64, Kdim);
        __builtin_amdgcn_s_wait_tensorcnt((short)2);   // current stage done
      } else {
        __builtin_amdgcn_s_wait_tensorcnt((short)0);
      }
    }
    __syncthreads();

    #pragma unroll
    for (int c = 0; c < 2; ++c) {                      // two K=32 chunks
      const int ko = c * 32;
      v16bf a[2], b[2];
      #pragma unroll
      for (int s = 0; s < 2; ++s) {
        const int ar = wm + s*16 + hl;                 // A: lane-major M
        v8bf a0 = *(const v8bf*)&sA[buf][ar][ko + g*8];        // K = 8g..8g+7
        v8bf a1 = *(const v8bf*)&sA[buf][ar][ko + 16 + g*8];   // K = 16+8g..
        a[s] = __builtin_shufflevector(a0, a1, 0,1,2,3,4,5,6,7,8,9,10,11,12,13,14,15);
        const int br = wn + s*16 + hl;                 // B: N = row of W
        v8bf b0 = *(const v8bf*)&sB[buf][br][ko + g*8];        // K = 8g..8g+7
        v8bf b1 = *(const v8bf*)&sB[buf][br][ko + 16 + g*8];   // K = 16+8g..
        b[s] = __builtin_shufflevector(b0, b1, 0,1,2,3,4,5,6,7,8,9,10,11,12,13,14,15);
      }
      #pragma unroll
      for (int i = 0; i < 2; ++i)
        #pragma unroll
        for (int j = 0; j < 2; ++j)
          acc[i][j] = __builtin_amdgcn_wmma_f32_16x16x32_bf16(
              false, a[i], false, b[j], (short)0, acc[i][j], false, false);
    }
    __syncthreads();
  }

  // ---- epilogue: bias + activation, f32 store ----
  #pragma unroll
  for (int si = 0; si < 2; ++si) {
    #pragma unroll
    for (int sj = 0; sj < 2; ++sj) {
      const int col = Nbase + wn + sj*16 + hl;
      const float bv = bias[col];
      #pragma unroll
      for (int r = 0; r < 8; ++r) {
        const int row = Mbase + wm + si*16 + 8*g + r;
        float val = acc[si][sj][r] + bv;
        if (act == 1) val = val * sigm(val);
        Y[(size_t)row * N + col] = val;
      }
    }
  }
}

// ---------------------------------------------------------------------------
// beta / fast_decay / slow_decay / fast_gate / slow_gate heads (f32, tiny)
// ---------------------------------------------------------------------------
__global__ __launch_bounds__(256)
void small_proj(const float* __restrict__ hs,
                const float* __restrict__ Wb,  const float* __restrict__ bb,
                const float* __restrict__ Wfd, const float* __restrict__ bfd,
                const float* __restrict__ fdb,
                const float* __restrict__ Wsd, const float* __restrict__ bsd,
                const float* __restrict__ sdb,
                const float* __restrict__ Wfg, const float* __restrict__ bfg,
                const float* __restrict__ Wsg, const float* __restrict__ bsg,
                float* __restrict__ BETA, float* __restrict__ FD,
                float* __restrict__ SD,   float* __restrict__ FG,
                float* __restrict__ SG)
{
  const int idx = blockIdx.x * blockDim.x + threadIdx.x;
  if (idx >= TOK * 5 * NHD) return;
  const int tok = idx / (5 * NHD);
  const int r   = idx % (5 * NHD);
  const int m   = r / NHD;
  const int n   = r % NHD;

  const float* Wrow; float bias; float* out;
  switch (m) {
    case 0:  Wrow = Wb  + n*HIDD; bias = bb[n];            out = BETA; break;
    case 1:  Wrow = Wfd + n*HIDD; bias = bfd[n] + fdb[n];  out = FD;   break;
    case 2:  Wrow = Wsd + n*HIDD; bias = bsd[n] + sdb[n];  out = SD;   break;
    case 3:  Wrow = Wfg + n*HIDD; bias = bfg[n];           out = FG;   break;
    default: Wrow = Wsg + n*HIDD; bias = bsg[n];           out = SG;   break;
  }
  const float4* x = (const float4*)(hs + (size_t)tok * HIDD);
  const float4* w = (const float4*)Wrow;
  float s = 0.f;
  #pragma unroll 4
  for (int k = 0; k < HIDD/4; ++k) {
    float4 a = x[k], b2 = w[k];
    s += a.x*b2.x + a.y*b2.y + a.z*b2.z + a.w*b2.w;
  }
  out[tok * NHD + n] = sigm(s + bias);
}

// ---------------------------------------------------------------------------
// Dual decayed delta-rule scan; state S[2][64][64] in LDS.  Output written
// directly as bf16 so the Wo GEMM consumes it without a conversion pass.
// ---------------------------------------------------------------------------
__global__ __launch_bounds__(128)
void delta_scan(const float* __restrict__ Q, const float* __restrict__ Kp,
                const float* __restrict__ V,
                const float* __restrict__ BETA, const float* __restrict__ FD,
                const float* __restrict__ SD,   const float* __restrict__ FG,
                const float* __restrict__ SG,   unsigned short* __restrict__ O)
{
  __shared__ float S[2][DH][DH];
  __shared__ float skt[DH], svt[DH], sqt[DH];
  __shared__ float obuf[2][DH];

  const int tid  = threadIdx.x;
  const int half = tid >> 6;     // 0 = fast, 1 = slow
  const int v    = tid & 63;
  const int b    = blockIdx.x >> 4;
  const int h    = blockIdx.x & 15;

  for (int i = tid; i < 2*DH*DH; i += 128) ((float*)S)[i] = 0.f;
  __syncthreads();

  for (int t = 0; t < LL; ++t) {
    const size_t tok = (size_t)(b*LL + t);
    if (tid < DH) {
      const float beta = BETA[tok*NHD + h];
      const size_t base = tok*HIDD + h*DH + tid;
      skt[tid] = Kp[base] * beta;
      svt[tid] = V[base]  * beta;
      sqt[tid] = Q[base];
    }
    __syncthreads();

    float kk = 0.f, vv = 0.f, kv = 0.f;
    #pragma unroll 8
    for (int k = 0; k < DH; ++k) { float a = skt[k], c = svt[k]; kk += a*a; vv += c*c; kv += a*c; }
    const float psi = sigm(3.f * fabsf(kv / ((sqrtf(kk) + 1e-8f) * (sqrtf(vv) + 1e-8f))));

    const float d = (half == 0) ? FD[tok*NHD + h] * (1.f - 0.10f * psi)
                                : SD[tok*NHD + h] * (1.f - 0.05f * psi);

    float* Sv = &S[half][0][v];
    float old = 0.f;
    #pragma unroll 8
    for (int k = 0; k < DH; ++k) old += Sv[k*DH] * skt[k];
    const float delta = svt[v] - old;

    float o = 0.f;
    #pragma unroll 8
    for (int k = 0; k < DH; ++k) {
      float s = Sv[k*DH];
      s = s * d + skt[k] * delta;
      Sv[k*DH] = s;
      o += s * sqt[k];
    }
    const float gate = (half == 0) ? FG[tok*NHD + h] : SG[tok*NHD + h];
    obuf[half][v] = o * gate;
    __syncthreads();

    if (tid < DH) {
      const float alpha = 0.5f + 0.3f * psi;
      O[tok*HIDD + h*DH + tid] =
          f2bfbits(alpha * obuf[0][tid] + (1.f - alpha) * obuf[1][tid]);
    }
    __syncthreads();
  }
}

// ---------------------------------------------------------------------------
extern "C" void kernel_launch(void* const* d_in, const int* in_sizes, int n_in,
                              void* d_out, int out_size, void* d_ws, size_t ws_size,
                              hipStream_t stream)
{
  const float* hs  = (const float*)d_in[0];
  const float* Wq  = (const float*)d_in[1];  const float* bq  = (const float*)d_in[2];
  const float* Wk  = (const float*)d_in[3];  const float* bk  = (const float*)d_in[4];
  const float* Wv  = (const float*)d_in[5];  const float* bv  = (const float*)d_in[6];
  const float* Wo  = (const float*)d_in[7];  const float* bo  = (const float*)d_in[8];
  const float* Wb  = (const float*)d_in[9];  const float* bb  = (const float*)d_in[10];
  const float* Wfd = (const float*)d_in[11]; const float* bfd = (const float*)d_in[12];
  const float* fdb = (const float*)d_in[13];
  const float* Wsd = (const float*)d_in[14]; const float* bsd = (const float*)d_in[15];
  const float* sdb = (const float*)d_in[16];
  const float* Wfg = (const float*)d_in[17]; const float* bfg = (const float*)d_in[18];
  const float* Wsg = (const float*)d_in[19]; const float* bsg = (const float*)d_in[20];

  // ---- workspace carve ----
  float* fws   = (float*)d_ws;
  float* Qb    = fws;                              // TOK*HIDD f32
  float* Kb    = Qb + (size_t)TOK * HIDD;
  float* Vb    = Kb + (size_t)TOK * HIDD;
  float* BETAp = Vb + (size_t)TOK * HIDD;
  float* FDp   = BETAp + (size_t)TOK * NHD;
  float* SDp   = FDp   + (size_t)TOK * NHD;
  float* FGp   = SDp   + (size_t)TOK * NHD;
  float* SGp   = FGp   + (size_t)TOK * NHD;
  unsigned short* hsb = (unsigned short*)(SGp + (size_t)TOK * NHD);
  unsigned short* Wqb = hsb + (size_t)TOK  * HIDD;
  unsigned short* Wkb = Wqb + (size_t)HIDD * HIDD;
  unsigned short* Wvb = Wkb + (size_t)HIDD * HIDD;
  unsigned short* Wob = Wvb + (size_t)HIDD * HIDD;
  unsigned short* Obb = Wob + (size_t)HIDD * HIDD;

  // ---- 1) one-shot bf16 conversion of activations + weights ----
  const int nhs4 = TOK  * HIDD / 4;
  const int nw4  = HIDD * HIDD / 4;
  to_bf16<<<(nhs4 + 255)/256, 256, 0, stream>>>(hs, hsb, nhs4);
  to_bf16<<<(nw4  + 255)/256, 256, 0, stream>>>(Wq, Wqb, nw4);
  to_bf16<<<(nw4  + 255)/256, 256, 0, stream>>>(Wk, Wkb, nw4);
  to_bf16<<<(nw4  + 255)/256, 256, 0, stream>>>(Wv, Wvb, nw4);
  to_bf16<<<(nw4  + 255)/256, 256, 0, stream>>>(Wo, Wob, nw4);

  // ---- 2) q/k/v projections (TDM -> LDS -> WMMA), silu epilogue ----
  dim3 gg(HIDD / 64, TOK / 64);
  gemm_bf16_wmma<<<gg, 128, 0, stream>>>(hsb, Wqb, bq, Qb, TOK, HIDD, HIDD, 1);
  gemm_bf16_wmma<<<gg, 128, 0, stream>>>(hsb, Wkb, bk, Kb, TOK, HIDD, HIDD, 1);
  gemm_bf16_wmma<<<gg, 128, 0, stream>>>(hsb, Wvb, bv, Vb, TOK, HIDD, HIDD, 1);

  // ---- 3) gating heads ----
  const int np = TOK * 5 * NHD;
  small_proj<<<(np + 255)/256, 256, 0, stream>>>(
      hs, Wb, bb, Wfd, bfd, fdb, Wsd, bsd, sdb, Wfg, bfg, Wsg, bsg,
      BETAp, FDp, SDp, FGp, SGp);

  // ---- 4) sequential dual delta-rule scan (writes bf16) ----
  delta_scan<<<BB * NHD, 128, 0, stream>>>(Qb, Kb, Vb, BETAp, FDp, SDp, FGp, SGp, Obb);

  // ---- 5) output projection ----
  gemm_bf16_wmma<<<gg, 128, 0, stream>>>(Obb, Wob, bo, (float*)d_out, TOK, HIDD, HIDD, 0);
}
